// CausalSelfAttention_4226247819618
// MI455X (gfx1250) — matrix operand order repaired and checked by reading, not compiled, not run on hardware
//
#include <hip/hip_runtime.h>

// ---- CDNA5 wave32 WMMA fragment types -------------------------------------
typedef __attribute__((ext_vector_type(16))) _Float16 v16h;
typedef __attribute__((ext_vector_type(8)))  _Float16 v8h;
typedef __attribute__((ext_vector_type(8)))  float    v8f;

#define DEV static __device__ __forceinline__

DEV v16h cat16(v8h lo, v8h hi) {
  return __builtin_shufflevector(lo, hi, 0,1,2,3,4,5,6,7,8,9,10,11,12,13,14,15);
}
// A-fragment (16x32 f16, MxK): lane L: row = L%16, elements 0..15 = K base+half*16 .. +15
// -> two contiguous b128 loads.
DEV v16h load_a(const _Float16* p) {           // p = row*ld + kb + half*16
  return cat16(*(const v8h*)p, *(const v8h*)(p + 8));
}
// B-fragment (32x16 f16, KxN): lane L: col = L%16, elements 0..15 = K base+half*16 .. +15
DEV v16h load_b(const _Float16* p) {           // p = col*ld + kb + half*16
  return cat16(*(const v8h*)p, *(const v8h*)(p + 8));
}
DEV v8f wmma(v16h a, v16h b, v8f c) {
  return __builtin_amdgcn_wmma_f32_16x16x32_f16(false, a, false, b, (short)0, c, false, false);
}

// ---- problem constants -----------------------------------------------------
#define BATCH 2
#define SEQ   2048
#define EMB   1024
#define NH    16
#define HD    64
#define ROWS  (BATCH*SEQ)          // 4096

// ---- precision-cast helpers ------------------------------------------------
__global__ void cast_f16_kernel(const float* __restrict__ in, _Float16* __restrict__ out, int n) {
  int i = blockIdx.x * blockDim.x + threadIdx.x;
  if (i < n) out[i] = (_Float16)in[i];
}
__global__ void transpose_cast_kernel(const float* __restrict__ in, _Float16* __restrict__ out,
                                      int K, int N) {     // in[K][N] -> out[N][K]
  int i = blockIdx.x * blockDim.x + threadIdx.x;
  if (i < K * N) {
    int k = i / N, n = i - k * N;
    out[n * K + k] = (_Float16)in[i];
  }
}

// ---- kernel 1: QKV GEMM  (x16[4096,1024] @ Wattn) --------------------------
// 4 waves/block, wave tile = 32m x 64n, K-step 32.  Epilogue scatters straight
// into q16/k16 [B,H,S,64] (q pre-scaled by 1/sqrt(hd)) and v16t [B,H,64,S].
__global__ __launch_bounds__(128)
void qkv_gemm_kernel(const _Float16* __restrict__ x16, const _Float16* __restrict__ wat,
                     const float* __restrict__ bias,
                     _Float16* __restrict__ q16, _Float16* __restrict__ k16,
                     _Float16* __restrict__ v16t) {
  const int lane = threadIdx.x & 31;
  const int wave = threadIdx.x >> 5;
  const int l16  = lane & 15, half = lane >> 4;
  const int m0 = blockIdx.y * 32;
  const int n0 = blockIdx.x * 256 + wave * 64;

  v8f acc[2][4];
  const v8f zf = {0.f,0.f,0.f,0.f,0.f,0.f,0.f,0.f};
  #pragma unroll
  for (int i = 0; i < 2; ++i)
    #pragma unroll
    for (int j = 0; j < 4; ++j) acc[i][j] = zf;

  const _Float16* pa0 = x16 + (m0 + l16) * EMB + half * 16;
  const _Float16* pa1 = pa0 + 16 * EMB;
  const _Float16* pb0 = wat + (n0 + l16) * EMB + half * 16;

  for (int kb = 0; kb < EMB; kb += 32) {
    v16h a0 = load_a(pa0 + kb);
    v16h a1 = load_a(pa1 + kb);
    #pragma unroll
    for (int nt = 0; nt < 4; ++nt) {
      v16h b = load_b(pb0 + nt * 16 * EMB + kb);
      acc[0][nt] = wmma(a0, b, acc[0][nt]);
      acc[1][nt] = wmma(a1, b, acc[1][nt]);
    }
  }

  #pragma unroll
  for (int nt = 0; nt < 4; ++nt) {
    const int n = n0 + nt * 16 + l16;
    const float bv = bias[n];
    const int region = n >> 10;          // 0=q 1=k 2=v
    const int col = n & 1023;
    const int h = col >> 6, d = col & 63;
    #pragma unroll
    for (int mt = 0; mt < 2; ++mt) {
      #pragma unroll
      for (int v = 0; v < 8; ++v) {
        const int r = m0 + mt * 16 + v + half * 8;   // global row in [0,4096)
        const int bb = r >> 11;                      // batch
        const int s  = r & (SEQ - 1);
        const float val = acc[mt][nt][v] + bv;
        if (region == 0)
          q16[(((bb * NH + h) * SEQ) + s) * HD + d] = (_Float16)(val * 0.125f);
        else if (region == 1)
          k16[(((bb * NH + h) * SEQ) + s) * HD + d] = (_Float16)val;
        else
          v16t[(((bb * NH + h) * HD) + d) * SEQ + s] = (_Float16)val;
      }
    }
  }
}

// ---- kernel 2: flash attention (causal) ------------------------------------
// One wave owns a 16-row query strip; waves are fully independent (no block
// barriers -> safe with divergent causal trip counts). Per 32-key block:
// 4 WMMAs for S = Q.K^T, online softmax in fp32 (shfl-xor reductions within
// 16-lane halves), P bounced C-layout -> A-layout via per-wave LDS, then
// 4 WMMAs for O += P.V.  Writes y16 in [B,S,H*64] layout for the proj GEMM.
__global__ __launch_bounds__(128)
void attn_kernel(const _Float16* __restrict__ q16, const _Float16* __restrict__ k16,
                 const _Float16* __restrict__ v16t, _Float16* __restrict__ y16) {
  __shared__ _Float16 Plds[4][16 * 32];
  const int lane = threadIdx.x & 31;
  const int wave = threadIdx.x >> 5;
  const int l16  = lane & 15, half = lane >> 4;
  const int bh = blockIdx.y;                      // 0..B*NH-1
  const int qs = blockIdx.x * 64 + wave * 16;     // query strip start

  const _Float16* Q  = q16 + (size_t)bh * SEQ * HD;
  const _Float16* Kp = k16 + (size_t)bh * SEQ * HD;
  const _Float16* Vt = v16t + (size_t)bh * HD * SEQ;

  const v16h aQ0 = load_a(Q + (qs + l16) * HD + half * 16);
  const v16h aQ1 = load_a(Q + (qs + l16) * HD + 32 + half * 16);

  const v8f zf = {0.f,0.f,0.f,0.f,0.f,0.f,0.f,0.f};
  v8f o[4] = {zf, zf, zf, zf};
  float mrow[8], lrow[8];
  #pragma unroll
  for (int v = 0; v < 8; ++v) { mrow[v] = -1e30f; lrow[v] = 0.f; }

  for (int j0 = 0; j0 <= qs + 15; j0 += 32) {
    // ---- S = Q K^T for a 16q x 32k block (two 16x16 C tiles)
    v8f s0 = zf, s1 = zf;
    {
      v16h b00 = load_b(Kp + (j0 + l16) * HD + half * 16);
      v16h b10 = load_b(Kp + (j0 + 16 + l16) * HD + half * 16);
      s0 = wmma(aQ0, b00, s0);
      s1 = wmma(aQ0, b10, s1);
      v16h b01 = load_b(Kp + (j0 + l16) * HD + 32 + half * 16);
      v16h b11 = load_b(Kp + (j0 + 16 + l16) * HD + 32 + half * 16);
      s0 = wmma(aQ1, b01, s0);
      s1 = wmma(aQ1, b11, s1);
    }
    // ---- causal mask + online softmax (scale folded into q16)
    const int key0 = j0 + l16, key1 = j0 + 16 + l16;
    float p0[8], p1[8], alpha[8];
    #pragma unroll
    for (int v = 0; v < 8; ++v) {
      const int row = qs + v + half * 8;
      float a = (key0 <= row) ? s0[v] : -1e30f;
      float b = (key1 <= row) ? s1[v] : -1e30f;
      float m = fmaxf(a, b);
      #pragma unroll
      for (int off = 8; off; off >>= 1) m = fmaxf(m, __shfl_xor(m, off, 16));
      const float mn = fmaxf(mrow[v], m);
      const float al = __expf(mrow[v] - mn);
      mrow[v] = mn;
      const float e0 = __expf(a - mn), e1 = __expf(b - mn);
      p0[v] = e0; p1[v] = e1;
      float ps = e0 + e1;
      #pragma unroll
      for (int off = 8; off; off >>= 1) ps += __shfl_xor(ps, off, 16);
      lrow[v] = lrow[v] * al + ps;
      alpha[v] = al;
    }
    #pragma unroll
    for (int dt = 0; dt < 4; ++dt)
      #pragma unroll
      for (int v = 0; v < 8; ++v) o[dt][v] *= alpha[v];

    // ---- C-layout P -> A-layout via LDS (wave-private, ds ops are in-order)
    _Float16* P = &Plds[wave][0];
    #pragma unroll
    for (int v = 0; v < 8; ++v) {
      P[(v + half * 8) * 32 + l16]      = (_Float16)p0[v];
      P[(v + half * 8) * 32 + 16 + l16] = (_Float16)p1[v];
    }
    const v16h aP = cat16(*(const v8h*)(P + l16 * 32 + half * 16),
                          *(const v8h*)(P + l16 * 32 + 8 + half * 16));
    // ---- O += P V  (V pre-transposed -> contiguous B-fragments)
    #pragma unroll
    for (int dt = 0; dt < 4; ++dt) {
      v16h b = load_b(Vt + (dt * 16 + l16) * SEQ + j0 + half * 16);
      o[dt] = wmma(aP, b, o[dt]);
    }
  }

  const int bb = bh >> 4, h = bh & 15;
  #pragma unroll
  for (int dt = 0; dt < 4; ++dt)
    #pragma unroll
    for (int v = 0; v < 8; ++v) {
      const int row = qs + v + half * 8;
      y16[((size_t)(bb * SEQ + row) * EMB) + h * HD + dt * 16 + l16] =
          (_Float16)(o[dt][v] / lrow[v]);
    }
}

// ---- kernel 3: output projection (y16 @ Wproj + b) -> fp32 -----------------
__global__ __launch_bounds__(128)
void proj_gemm_kernel(const _Float16* __restrict__ y16, const _Float16* __restrict__ wpt,
                      const float* __restrict__ bias, float* __restrict__ out) {
  const int lane = threadIdx.x & 31;
  const int wave = threadIdx.x >> 5;
  const int l16  = lane & 15, half = lane >> 4;
  const int m0 = blockIdx.y * 32;
  const int n0 = blockIdx.x * 256 + wave * 64;

  v8f acc[2][4];
  const v8f zf = {0.f,0.f,0.f,0.f,0.f,0.f,0.f,0.f};
  #pragma unroll
  for (int i = 0; i < 2; ++i)
    #pragma unroll
    for (int j = 0; j < 4; ++j) acc[i][j] = zf;

  const _Float16* pa0 = y16 + (m0 + l16) * EMB + half * 16;
  const _Float16* pa1 = pa0 + 16 * EMB;
  const _Float16* pb0 = wpt + (n0 + l16) * EMB + half * 16;

  for (int kb = 0; kb < EMB; kb += 32) {
    v16h a0 = load_a(pa0 + kb);
    v16h a1 = load_a(pa1 + kb);
    #pragma unroll
    for (int nt = 0; nt < 4; ++nt) {
      v16h b = load_b(pb0 + nt * 16 * EMB + kb);
      acc[0][nt] = wmma(a0, b, acc[0][nt]);
      acc[1][nt] = wmma(a1, b, acc[1][nt]);
    }
  }

  #pragma unroll
  for (int nt = 0; nt < 4; ++nt) {
    const int n = n0 + nt * 16 + l16;
    const float bv = bias[n];
    #pragma unroll
    for (int mt = 0; mt < 2; ++mt) {
      #pragma unroll
      for (int v = 0; v < 8; ++v) {
        const int r = m0 + mt * 16 + v + half * 8;
        out[(size_t)r * EMB + n] = acc[mt][nt][v] + bv;
      }
    }
  }
}

// ---- host-side orchestration ----------------------------------------------
extern "C" void kernel_launch(void* const* d_in, const int* in_sizes, int n_in,
                              void* d_out, int out_size, void* d_ws, size_t ws_size,
                              hipStream_t stream) {
  const float* x      = (const float*)d_in[0];   // [2,2048,1024]
  const float* W_attn = (const float*)d_in[1];   // [1024,3072]
  const float* b_attn = (const float*)d_in[2];   // [3072]
  const float* W_proj = (const float*)d_in[3];   // [1024,1024]
  const float* b_proj = (const float*)d_in[4];   // [1024]
  float* out = (float*)d_out;                    // [2,2048,1024]

  char* ws = (char*)d_ws;
  _Float16* x16   = (_Float16*)(ws);                     // 4096*1024      (8 MB)
  _Float16* wa16t = (_Float16*)(ws + (8u  << 20));       // 3072*1024      (6 MB)
  _Float16* wp16t = (_Float16*)(ws + (14u << 20));       // 1024*1024      (2 MB)
  _Float16* q16   = (_Float16*)(ws + (16u << 20));       // [B,H,S,64]     (8 MB)
  _Float16* k16   = (_Float16*)(ws + (24u << 20));       // [B,H,S,64]     (8 MB)
  _Float16* v16t  = (_Float16*)(ws + (32u << 20));       // [B,H,64,S]     (8 MB)
  _Float16* y16   = (_Float16*)(ws + (40u << 20));       // [B,S,1024]     (8 MB)

  // precision casts (one-time, bandwidth-trivial; everything stays L2-resident)
  cast_f16_kernel<<<dim3(ROWS * EMB / 256), dim3(256), 0, stream>>>(x, x16, ROWS * EMB);
  transpose_cast_kernel<<<dim3(EMB * 3 * EMB / 256), dim3(256), 0, stream>>>(W_attn, wa16t, EMB, 3 * EMB);
  transpose_cast_kernel<<<dim3(EMB * EMB / 256), dim3(256), 0, stream>>>(W_proj, wp16t, EMB, EMB);

  // QKV projection: M=4096, N=3072, K=1024; block tile 32x256 (4 waves)
  qkv_gemm_kernel<<<dim3(3 * EMB / 256, ROWS / 32), dim3(128), 0, stream>>>(
      x16, wa16t, b_attn, q16, k16, v16t);

  // flash attention: grid = (S/64 q-blocks, B*H heads), 4 independent waves/block
  attn_kernel<<<dim3(SEQ / 64, BATCH * NH), dim3(128), 0, stream>>>(q16, k16, v16t, y16);

  // output projection: M=4096, N=1024, K=1024
  proj_gemm_kernel<<<dim3(EMB / 256, ROWS / 32), dim3(128), 0, stream>>>(
      y16, wp16t, b_proj, out);
}
